// MultiHeadAttention_71459665871450
// MI455X (gfx1250) — compile-verified, matrix operand order checked by reading, not run on hardware
//
#include <hip/hip_runtime.h>
#include <hip/hip_bf16.h>
#include <stdint.h>

typedef _Float16 half_t;
typedef __attribute__((ext_vector_type(16))) _Float16 v16h;
typedef __attribute__((ext_vector_type(8)))  _Float16 v8h;
typedef __attribute__((ext_vector_type(8)))  float    v8f;
typedef __attribute__((ext_vector_type(4)))  unsigned int u32x4;
typedef __attribute__((ext_vector_type(8)))  int i32x8;
typedef __attribute__((ext_vector_type(4)))  int i32x4;

#define B_ 2
#define S_ 2048
#define E_ 1024
#define H_ 16
#define D_ 64
#define M_ (B_*S_)

#if __has_builtin(__builtin_amdgcn_tensor_load_to_lds)
#define HAS_TDM 1
#endif
#if __has_include(<hip/amd_detail/amd_gfx1250_TDM.h>)
#define TDM_6ARG 1
#endif

// ---------------------------------------------------------------------------
// WMMA fragment helpers (CDNA5 16-bit layout, wave32): v16h = two 16B chunks
// at +hi*8 and +16+hi*8 halfs; lane%16 = M row (A) / N col (B).
// ---------------------------------------------------------------------------
static __device__ __forceinline__ v16h pack16(v8h lo, v8h hi8) {
  v16h r;
#pragma unroll
  for (int i = 0; i < 8; ++i) { r[i] = lo[i]; r[i + 8] = hi8[i]; }
  return r;
}

static __device__ __forceinline__ v16h frag_rm(const half_t* rowptr, int kbase, int hi) {
  v8h a = *(const v8h*)(rowptr + kbase + hi * 8);
  v8h b = *(const v8h*)(rowptr + kbase + 16 + hi * 8);
  return pack16(a, b);
}

static __device__ __forceinline__ v8f wmma16(v16h a, v16h b, v8f c) {
  return __builtin_amdgcn_wmma_f32_16x16x32_f16(false, a, false, b, (short)0, c,
                                                false, false);
}

// ---------------------------------------------------------------------------
// TDM: 2D f16 tile (rows x row_len elems, global row stride gstride elems)
// -> LDS at lds_off, with D# LDS padding pad_amount(+1) DWORDs every
// 2^(pad_interval+1) DWORDs.  D# fields per CDNA5 ISA 8.3/8.4.
// ---------------------------------------------------------------------------
#if defined(HAS_TDM)
static __device__ __forceinline__ void tdm_load_2d_f16(
    const half_t* gsrc, uint32_t lds_off, int rows, int row_len, int gstride,
    uint32_t pad_interval_code, uint32_t pad_amount_code) {
  const uint64_t ga = (uint64_t)(uintptr_t)gsrc;
  u32x4 g0;
  g0[0] = 1u;                                   // count=1 user descriptor
  g0[1] = lds_off;                              // lds_addr (bytes)
  g0[2] = (uint32_t)ga;                         // global_addr[31:0]
  g0[3] = (uint32_t)((ga >> 32) & 0x1FFFFFFu)   // global_addr[56:32]
        | (2u << 30);                           // type=2 ("image")
  const uint32_t dim0 = 1u << 30, dim1 = 1u << 30;  // huge: no OOB clamping
  i32x8 g1;
  g1[0] = (int)((1u << 16)                      // data_size = 2 bytes
              | (1u << 20)                      // pad_enable
              | (pad_interval_code << 22) | (pad_amount_code << 25));
  g1[1] = (int)((dim0 & 0xFFFFu) << 16);        // atomic_addr=0 | dim0 lo16
  g1[2] = (int)(((dim0 >> 16) & 0xFFFFu) | ((dim1 & 0xFFFFu) << 16));
  g1[3] = (int)(((dim1 >> 16) & 0xFFFFu) | ((uint32_t)row_len << 16)); // tile_dim0
  g1[4] = (int)((uint32_t)rows & 0xFFFFu);      // tile_dim1 | tile_dim2=0
  g1[5] = (int)(uint32_t)gstride;               // tensor_dim0_stride lo32
  g1[6] = 0;
  g1[7] = 0;
  const i32x4 gz = {0, 0, 0, 0};
#if defined(TDM_6ARG)
  const i32x8 gz8 = {0, 0, 0, 0, 0, 0, 0, 0};
  __builtin_amdgcn_tensor_load_to_lds(g0, g1, gz, gz, gz8, 0);
#else
  __builtin_amdgcn_tensor_load_to_lds(g0, g1, gz, gz, 0);
#endif
}
#endif

// ---------------------------------------------------------------------------
// Y = X @ W^T + bias  (X [4096x1024] f32, W [1024x1024] f32 row-major).
// WG tile 128x128, 8 waves (4Mx2N), wave tile 32x64 -> 8 WMMA per K-step(32).
// MODE 0: f16 [B,H,S,D]; MODE 1: f16 [B,H,D,S] (V transposed); MODE 2: f32.
// ---------------------------------------------------------------------------
template <int MODE>
__global__ __launch_bounds__(256) void gemm_bias_kernel(
    const float* __restrict__ X, const float* __restrict__ W,
    const float* __restrict__ bias, half_t* __restrict__ outh,
    float* __restrict__ outf) {
  constexpr int LDA = 40, LDB = 40;   // 32+8 halfs: conflict-free b128 reads
  __shared__ half_t a_lds[128 * LDA];
  __shared__ half_t b_lds[128 * LDB];

  const int tid = threadIdx.x;
  const int lane = tid & 31, wave = tid >> 5;
  const int hi = lane >> 4, lm = lane & 15;
  const int wm = wave >> 1, wn = wave & 1;
  const int mbase = blockIdx.x * 128;
  const int nbase = blockIdx.y * 128;
  const int srow = tid >> 1, sseg = tid & 1;   // staging: 128 rows x (2x16)

  v8f acc[2][4] = {};

  for (int k0 = 0; k0 < E_; k0 += 32) {
    {  // stage X and W tiles, f32 -> f16
      const float* asrc = X + (size_t)(mbase + srow) * E_ + k0 + sseg * 16;
      const float* bsrc = W + (size_t)(nbase + srow) * E_ + k0 + sseg * 16;
      half_t* adst = &a_lds[srow * LDA + sseg * 16];
      half_t* bdst = &b_lds[srow * LDB + sseg * 16];
#pragma unroll
      for (int i = 0; i < 16; i += 4) {
        float4 va = *(const float4*)(asrc + i);
        float4 vb = *(const float4*)(bsrc + i);
        adst[i + 0] = (half_t)va.x; adst[i + 1] = (half_t)va.y;
        adst[i + 2] = (half_t)va.z; adst[i + 3] = (half_t)va.w;
        bdst[i + 0] = (half_t)vb.x; bdst[i + 1] = (half_t)vb.y;
        bdst[i + 2] = (half_t)vb.z; bdst[i + 3] = (half_t)vb.w;
      }
      if (k0 + 32 < E_) {  // gfx1250 global_prefetch_b8 hints
        __builtin_prefetch(asrc + 32, 0, 1);
        __builtin_prefetch(bsrc + 32, 0, 1);
      }
    }
    __syncthreads();

    // hoist all LDS fragment loads, then a back-to-back WMMA burst
    v16h a0 = frag_rm(&a_lds[(wm * 32 + lm) * LDA], 0, hi);
    v16h a1 = frag_rm(&a_lds[(wm * 32 + 16 + lm) * LDA], 0, hi);
    v16h bf[4];
#pragma unroll
    for (int nf = 0; nf < 4; ++nf)
      bf[nf] = frag_rm(&b_lds[(wn * 64 + nf * 16 + lm) * LDB], 0, hi);
#pragma unroll
    for (int nf = 0; nf < 4; ++nf) {
      acc[0][nf] = wmma16(a0, bf[nf], acc[0][nf]);
      acc[1][nf] = wmma16(a1, bf[nf], acc[1][nf]);
    }
    __syncthreads();
  }

#pragma unroll
  for (int mf = 0; mf < 2; ++mf)
#pragma unroll
    for (int nf = 0; nf < 4; ++nf) {
      const int n = nbase + wn * 64 + nf * 16 + lm;
      const float bv = bias[n];
#pragma unroll
      for (int r = 0; r < 8; ++r) {
        const int m = mbase + wm * 32 + mf * 16 + r + 8 * hi;
        const float v = acc[mf][nf][r] + bv;
        if (MODE == 2) {
          outf[(size_t)m * E_ + n] = v;
        } else {
          const int batch = m / S_, s = m % S_;
          const int h = n / D_, d = n % D_;
          if (MODE == 0)
            outh[(((size_t)batch * H_ + h) * S_ + s) * D_ + d] = (half_t)v;
          else
            outh[(((size_t)batch * H_ + h) * D_ + d) * S_ + s] = (half_t)v;
        }
      }
    }
}

// ---------------------------------------------------------------------------
// Causal flash attention. Grid (S/128, B*H), 8 waves x 16 query rows.
// Uniform 64-key blocks; wave0 TDM-stages K(64x64) and Vt(64x64) tiles into
// double-buffered LDS (72-half padded stride via D# padding), TENSORcnt
// synchronized; per block: 8 WMMA QK^T + online softmax + 8 WMMA PV.
// All fragment loads of a phase are hoisted ahead of the WMMA burst so the
// ds_load_b128s pipeline instead of serializing on s_wait_dscnt 0.
// ---------------------------------------------------------------------------
__global__ __launch_bounds__(256) void flash_attn_kernel(
    const half_t* __restrict__ Qh, const half_t* __restrict__ Kh,
    const half_t* __restrict__ Vt, float* __restrict__ attn) {
  constexpr int LDK = 72;                       // 64 + 4-DWORD pad per row
  __shared__ half_t k_lds[2][64 * LDK];         // 2 x 9216 B
  __shared__ half_t v_lds[2][64 * LDK];         // 2 x 9216 B
  __shared__ half_t p_lds[8][16 * LDK];         // per-wave P tile, padded

  const int tid = threadIdx.x;
  const int lane = tid & 31, wave = tid >> 5;
  const int hi = lane >> 4, lm = lane & 15;
  const int bh = blockIdx.y;
  const int batch = bh / H_, head = bh % H_;

  const half_t* qp = Qh + (size_t)bh * S_ * D_;
  const half_t* kp = Kh + (size_t)bh * S_ * D_;
  const half_t* vp = Vt + (size_t)bh * D_ * S_;  // [D,S]

  const int qr0 = blockIdx.x * 128 + wave * 16;
  const int nblk = (blockIdx.x + 1) * 2;         // uniform across the WG

  const v16h qa0 = frag_rm(qp + (qr0 + lm) * D_, 0, hi);
  const v16h qa1 = frag_rm(qp + (qr0 + lm) * D_, 32, hi);

  float mrow[8], lrow[8];
#pragma unroll
  for (int r = 0; r < 8; ++r) { mrow[r] = -1e30f; lrow[r] = 0.f; }
  v8f o[4] = {};

  half_t* pl = &p_lds[wave][0];
  const float scale = 0.125f;  // 1/sqrt(D)

#if defined(HAS_TDM)
  if (wave == 0) {  // prologue: stage block 0 into buffer 0
    tdm_load_2d_f16(kp, (uint32_t)(uintptr_t)&k_lds[0][0], 64, 64, D_, 4, 3);
    tdm_load_2d_f16(vp, (uint32_t)(uintptr_t)&v_lds[0][0], 64, 64, S_, 4, 3);
  }
#endif

  for (int ib = 0; ib < nblk; ++ib) {
    const int kb = ib * 64;
    const int buf = ib & 1;

#if defined(HAS_TDM)
    if (wave == 0) {
      if (ib + 1 < nblk) {  // overlap next block's DMA with this compute
        const int kn = (ib + 1) * 64;
        tdm_load_2d_f16(kp + (size_t)kn * D_,
                        (uint32_t)(uintptr_t)&k_lds[1 - buf][0], 64, 64, D_, 4, 3);
        tdm_load_2d_f16(vp + kn,
                        (uint32_t)(uintptr_t)&v_lds[1 - buf][0], 64, 64, S_, 4, 3);
        __builtin_amdgcn_s_wait_tensorcnt(2);   // current block's 2 ops done
      } else {
        __builtin_amdgcn_s_wait_tensorcnt(0);
      }
    }
    __syncthreads();
#else
    {  // fallback cooperative staging (64 rows x 4 segs x 16 halfs)
      const int row = tid >> 2, seg = tid & 3;
      const v8h* ks = (const v8h*)(kp + (size_t)(kb + row) * D_ + seg * 16);
      const v8h* vs = (const v8h*)(vp + (size_t)row * S_ + kb + seg * 16);
      v8h* kd = (v8h*)(&k_lds[buf][row * LDK + seg * 16]);
      v8h* vd = (v8h*)(&v_lds[buf][row * LDK + seg * 16]);
      kd[0] = ks[0]; kd[1] = ks[1];
      vd[0] = vs[0]; vd[1] = vs[1];
      __syncthreads();
    }
#endif

    if (kb <= qr0 + 15) {  // this wave has unmasked keys in the block
      // ---- hoisted K fragment loads (8x ds_load_b128 pairs pipeline) ----
      v16h kf[4][2];
#pragma unroll
      for (int nf = 0; nf < 4; ++nf) {
        const half_t* krow = &k_lds[buf][(nf * 16 + lm) * LDK];
        kf[nf][0] = frag_rm(krow, 0, hi);
        kf[nf][1] = frag_rm(krow, 32, hi);
      }
      // ---- scores 16x64: back-to-back WMMA burst ----
      v8f sf[4];
#pragma unroll
      for (int nf = 0; nf < 4; ++nf) {
        v8f c = {};
        c = wmma16(qa0, kf[nf][0], c);
        c = wmma16(qa1, kf[nf][1], c);
        sf[nf] = c;
      }

      // ---- scale + causal mask + online softmax ----
      float pnew[4][8];
#pragma unroll
      for (int r = 0; r < 8; ++r) {
        const int row = qr0 + r + 8 * hi;
        float mx = -1e30f;
#pragma unroll
        for (int nf = 0; nf < 4; ++nf) {
          const int col = kb + nf * 16 + lm;
          const float v = sf[nf][r] * scale;
          pnew[nf][r] = (col <= row) ? v : -1e30f;
          mx = fmaxf(mx, pnew[nf][r]);
        }
#pragma unroll
        for (int xm = 1; xm <= 8; xm <<= 1)  // xor<16: within lane-half
          mx = fmaxf(mx, __shfl_xor(mx, xm));
        const float mnew = fmaxf(mrow[r], mx);
        const float fac = __expf(mrow[r] - mnew);
        float rs = 0.f;
#pragma unroll
        for (int nf = 0; nf < 4; ++nf) {
          const float p = __expf(pnew[nf][r] - mnew);
          pnew[nf][r] = p;
          rs += p;
        }
#pragma unroll
        for (int xm = 1; xm <= 8; xm <<= 1)
          rs += __shfl_xor(rs, xm);
        lrow[r] = lrow[r] * fac + rs;
        mrow[r] = mnew;
#pragma unroll
        for (int df = 0; df < 4; ++df) o[df][r] *= fac;
      }

      // ---- P: C-layout -> f16 A-layout via per-wave LDS tile ----
#pragma unroll
      for (int nf = 0; nf < 4; ++nf)
#pragma unroll
        for (int r = 0; r < 8; ++r)
          pl[(r + 8 * hi) * LDK + nf * 16 + lm] = (half_t)pnew[nf][r];
      asm volatile("s_wait_dscnt 0" ::: "memory");  // same-wave DS RAW
      const v16h pa0 = frag_rm(pl + lm * LDK, 0, hi);
      const v16h pa1 = frag_rm(pl + lm * LDK, 32, hi);

      // ---- hoisted V fragment loads, then PV WMMA burst ----
      v16h vf[4][2];
#pragma unroll
      for (int df = 0; df < 4; ++df) {
        const half_t* vrow = &v_lds[buf][(df * 16 + lm) * LDK];
        vf[df][0] = frag_rm(vrow, 0, hi);
        vf[df][1] = frag_rm(vrow, 32, hi);
      }
#pragma unroll
      for (int df = 0; df < 4; ++df) {
        o[df] = wmma16(pa0, vf[df][0], o[df]);
        o[df] = wmma16(pa1, vf[df][1], o[df]);
      }
    }
    __syncthreads();
  }

  // ---- normalize and store attn [B,S,E] f32 ----
#pragma unroll
  for (int df = 0; df < 4; ++df)
#pragma unroll
    for (int r = 0; r < 8; ++r) {
      const int row = qr0 + r + 8 * hi;
      const int col = head * D_ + df * 16 + lm;
      attn[((size_t)batch * S_ + row) * E_ + col] = o[df][r] / lrow[r];
    }
}

// ---------------------------------------------------------------------------
extern "C" void kernel_launch(void* const* d_in, const int* in_sizes, int n_in,
                              void* d_out, int out_size, void* d_ws,
                              size_t ws_size, hipStream_t stream) {
  const float* Q  = (const float*)d_in[0];
  const float* K  = (const float*)d_in[1];
  const float* V  = (const float*)d_in[2];
  const float* Wq = (const float*)d_in[3];
  const float* bq = (const float*)d_in[4];
  const float* Wk = (const float*)d_in[5];
  const float* bk = (const float*)d_in[6];
  const float* Wv = (const float*)d_in[7];
  const float* bv = (const float*)d_in[8];
  const float* Wo = (const float*)d_in[9];
  const float* bo = (const float*)d_in[10];
  float* out = (float*)d_out;

  const size_t nqkv = (size_t)B_ * H_ * S_ * D_;
  half_t* qb   = (half_t*)d_ws;            // 8 MB f16 [B,H,S,D]
  half_t* kb   = qb + nqkv;                // 8 MB f16 [B,H,S,D]
  half_t* vtb  = kb + nqkv;                // 8 MB f16 [B,H,D,S]
  float*  attn = (float*)(vtb + nqkv);     // 16 MB f32 [B,S,E]

  const dim3 blk(256);
  const dim3 gg(M_ / 128, E_ / 128);
  gemm_bias_kernel<0><<<gg, blk, 0, stream>>>(Q, Wq, bq, qb, nullptr);
  gemm_bias_kernel<0><<<gg, blk, 0, stream>>>(K, Wk, bk, kb, nullptr);
  gemm_bias_kernel<1><<<gg, blk, 0, stream>>>(V, Wv, bv, vtb, nullptr);
  flash_attn_kernel<<<dim3(S_ / 128, B_ * H_), blk, 0, stream>>>(qb, kb, vtb,
                                                                 attn);
  gemm_bias_kernel<2><<<gg, blk, 0, stream>>>(attn, Wo, bo, nullptr, out);
}
